// MHAttention_84791244357794
// MI455X (gfx1250) — matrix operand order repaired and checked by reading, not compiled, not run on hardware
//
#include <hip/hip_runtime.h>
#include <hip/hip_bf16.h>

typedef __attribute__((ext_vector_type(16))) _Float16 v16h;
typedef __attribute__((ext_vector_type(8)))  float    v8f;
typedef int v4i32 __attribute__((vector_size(16)));

#define TM 128
#define TKDIM 32
#define LDA_PAD 40
#define LDB_PAD 40

// ---- CDNA5 async global->LDS path (guarded; fallback = sync loads) ----
#if defined(__AMDGCN__) && defined(__has_builtin)
#if __has_builtin(__builtin_amdgcn_global_load_async_to_lds_b128)
#define HAS_ASYNC 1
#endif
#endif
#ifndef HAS_ASYNC
#define HAS_ASYNC 0
#endif

#if HAS_ASYNC
__device__ __forceinline__ void async_copy_b128(const void* g, void* l) {
    __builtin_amdgcn_global_load_async_to_lds_b128(
        (__attribute__((address_space(1))) v4i32*)g,
        (__attribute__((address_space(3))) v4i32*)l, 0, 0);
}
#if __has_builtin(__builtin_amdgcn_s_wait_asynccnt)
#define WAIT_ASYNC(n) __builtin_amdgcn_s_wait_asynccnt(n)
#else
#define WAIT_ASYNC(n) asm volatile("s_wait_asynccnt %0" ::"n"(n))
#endif
#else
#define WAIT_ASYNC(n) ((void)0)
#endif

__device__ __forceinline__ float gelu_exact(float x) {
    return 0.5f * x * (1.0f + erff(x * 0.70710678118654752f));
}

// C[m,n] = sum_k A[m,k] * B(k,n)   (f32 accumulate via v_wmma_f32_16x16x32_f16)
// BLAYOUT 0: B stored as Bt (N x Kdim, row-major)   -> B(k,n) = Bt[n*ldb + k]
// BLAYOUT 1: B stored as (Kdim x N, row-major)      -> B(k,n) = B[k*ldb + n]
// BIASMODE: 0 none, 1 bias[row], 2 bias[col]; ACT: 0 none, 1 exact GELU
template<int TN, int BLAYOUT, int BIASMODE, int ACT, typename TOUT>
__global__ __launch_bounds__(256) void gemm_wmma_kernel(
    const _Float16* __restrict__ A, int lda, long sA0, long sA1,
    const _Float16* __restrict__ Bm, int ldb, long sB0, long sB1,
    TOUT* __restrict__ Cm, int ldc, long sC0, long sC1,
    const float* __restrict__ bias, int Kdim, int zdiv)
{
    constexpr int WN    = TN / 2;   // wave N extent (waves arranged 4 x 2)
    constexpr int SUB_N = WN / 16;  // 2 or 4
    constexpr int SUB_M = 2;        // wave M extent = 32
    // async b128 copies issued per thread per tile (A: 2, BT: TN/64)
    constexpr int NA = 2 + (BLAYOUT == 0 ? TN / 64 : 0);

    __shared__ _Float16 ldsA[2][TM * LDA_PAD];
    __shared__ _Float16 ldsB[2][TN * LDB_PAD];

    const int tid   = threadIdx.x;
    const int lane  = tid & 31;
    const int wave  = tid >> 5;
    const int waveM = wave >> 1;    // 0..3
    const int waveN = wave & 1;     // 0..1
    const int m2    = lane & 15;
    const int hi    = lane >> 4;

    const int z  = blockIdx.z;
    const int zo = z / zdiv, zi = z % zdiv;
    const _Float16* Ab = A  + zo * sA0 + zi * sA1;
    const _Float16* Bb = Bm + zo * sB0 + zi * sB1;
    TOUT*           Cb = Cm + zo * sC0 + zi * sC1;

    const int blockM = blockIdx.y * TM;
    const int blockN = blockIdx.x * TN;

    v8f acc[SUB_M][SUB_N];
    #pragma unroll
    for (int i = 0; i < SUB_M; ++i)
        #pragma unroll
        for (int j = 0; j < SUB_N; ++j)
            #pragma unroll
            for (int r = 0; r < 8; ++r) acc[i][j][r] = 0.0f;

    // ---- stage one k-tile (A: 128x32, B -> LDS as [n][k]) into buffer `buf`
    // All trip counts are compile-time exact: every lane participates (no EXEC
    // divergence -> WMMA-legal, no saveexec in the hot loop).
    auto issue_tile = [&](int kt, int buf) {
        {
            int row = tid >> 1;
            int kc  = (tid & 1) << 4;
            const _Float16* g = Ab + (long)(blockM + row) * lda + kt + kc;
            _Float16* l = &ldsA[buf][row * LDA_PAD + kc];
#if HAS_ASYNC
            async_copy_b128(g, l);
            async_copy_b128(g + 8, l + 8);
#else
            reinterpret_cast<uint4*>(l)[0] = reinterpret_cast<const uint4*>(g)[0];
            reinterpret_cast<uint4*>(l)[1] = reinterpret_cast<const uint4*>(g)[1];
#endif
        }
        if (BLAYOUT == 0) {
            #pragma unroll
            for (int it = 0; it < TN / 64; ++it) {
                int idx = tid + it * 256;
                int row = idx >> 2;
                int kc  = (idx & 3) << 3;
                const _Float16* g = Bb + (long)(blockN + row) * ldb + kt + kc;
                _Float16* l = &ldsB[buf][row * LDB_PAD + kc];
#if HAS_ASYNC
                async_copy_b128(g, l);
#else
                *reinterpret_cast<uint4*>(l) = *reinterpret_cast<const uint4*>(g);
#endif
            }
        } else {
            #pragma unroll
            for (int it = 0; it < TN / 64; ++it) {
                int idx = (tid + it * 256) * 8;
                int kk = idx / TN;
                int nc = idx % TN;
                uint4 v = *reinterpret_cast<const uint4*>(
                    Bb + (long)(kt + kk) * ldb + blockN + nc);
                _Float16 tmp[8];
                *reinterpret_cast<uint4*>(tmp) = v;
                #pragma unroll
                for (int i = 0; i < 8; ++i)
                    ldsB[buf][(nc + i) * LDB_PAD + kk] = tmp[i];
            }
        }
    };

    // ---- consume one staged tile with WMMA (ISA 7.12.2 16-bit layouts) ----
    auto compute_tile = [&](int buf) {
        v16h afrag[SUB_M];
        #pragma unroll
        for (int mi = 0; mi < SUB_M; ++mi) {
            const _Float16* pa =
                &ldsA[buf][(waveM * 32 + mi * 16 + m2) * LDA_PAD + hi * 8];
            #pragma unroll
            for (int i = 0; i < 8; ++i) {
                afrag[mi][i]     = pa[i];
                afrag[mi][8 + i] = pa[16 + i];
            }
        }
        v16h bfrag[SUB_N];
        #pragma unroll
        for (int ni = 0; ni < SUB_N; ++ni) {
            const _Float16* pb =
                &ldsB[buf][(waveN * WN + ni * 16 + m2) * LDB_PAD + hi * 8];
            #pragma unroll
            for (int i = 0; i < 8; ++i) {
                bfrag[ni][i]     = pb[i];
                bfrag[ni][8 + i] = pb[16 + i];
            }
        }
        #pragma unroll
        for (int mi = 0; mi < SUB_M; ++mi)
            #pragma unroll
            for (int ni = 0; ni < SUB_N; ++ni)
                acc[mi][ni] = __builtin_amdgcn_wmma_f32_16x16x32_f16(
                    false, afrag[mi], false, bfrag[ni],
                    (short)0, acc[mi][ni], false, false);
    };

    const int nk = Kdim / TKDIM;
    issue_tile(0, 0);

    // steady state: branch-free double-buffered pipeline (last tile peeled)
    for (int t = 0; t < nk - 1; ++t) {
        const int buf = t & 1;
        __syncthreads();                    // everyone done with buf^1
        issue_tile((t + 1) * TKDIM, buf ^ 1);
        WAIT_ASYNC(NA);                     // tile t complete (in-order)
        __syncthreads();
        compute_tile(buf);
    }
    WAIT_ASYNC(0);
    __syncthreads();
    compute_tile((nk - 1) & 1);

    // ---- epilogue ----
    #pragma unroll
    for (int mi = 0; mi < SUB_M; ++mi) {
        #pragma unroll
        for (int ni = 0; ni < SUB_N; ++ni) {
            int col = blockN + waveN * WN + ni * 16 + m2;
            #pragma unroll
            for (int r = 0; r < 8; ++r) {
                int row = blockM + waveM * 32 + mi * 16 + hi * 8 + r;
                float v = acc[mi][ni][r];
                if (BIASMODE == 1) v += bias[row];
                else if (BIASMODE == 2) v += bias[col];
                if (ACT == 1) v = gelu_exact(v);
                Cb[(long)row * ldc + col] = (TOUT)v;
            }
        }
    }
}

// in-place softmax over rows of length 256 (f16), one wave per row
__global__ __launch_bounds__(256) void softmax256_kernel(_Float16* __restrict__ P,
                                                         float scale)
{
    long row  = (long)blockIdx.x * 8 + (threadIdx.x >> 5);
    int  lane = threadIdx.x & 31;
    _Float16* p = P + row * 256;
    float x[8];
    float mx = -3.0e38f;
    #pragma unroll
    for (int i = 0; i < 8; ++i) {
        x[i] = (float)p[lane + i * 32] * scale;
        mx = fmaxf(mx, x[i]);
    }
    #pragma unroll
    for (int off = 16; off > 0; off >>= 1) mx = fmaxf(mx, __shfl_xor(mx, off, 32));
    float s = 0.0f;
    #pragma unroll
    for (int i = 0; i < 8; ++i) { x[i] = __expf(x[i] - mx); s += x[i]; }
    #pragma unroll
    for (int off = 16; off > 0; off >>= 1) s += __shfl_xor(s, off, 32);
    float inv = 1.0f / s;
    #pragma unroll
    for (int i = 0; i < 8; ++i) p[lane + i * 32] = (_Float16)(x[i] * inv);
}

__global__ __launch_bounds__(256) void cvt_f32_to_f16(const float* __restrict__ s,
                                                      _Float16* __restrict__ d,
                                                      long n)
{
    long i = (long)blockIdx.x * blockDim.x + threadIdx.x;
    long stride = (long)gridDim.x * blockDim.x;
    for (; i < n; i += stride) d[i] = (_Float16)s[i];
}

extern "C" void kernel_launch(void* const* d_in, const int* in_sizes, int n_in,
                              void* d_out, int out_size, void* d_ws, size_t ws_size,
                              hipStream_t stream)
{
    const int Bn = 2, S = 4096, Cn = 1024, Hn = 16, Dn = 64, Kn = 256;

    const float* tensor = (const float*)d_in[0];
    const float* Wq     = (const float*)d_in[1];
    const float* Wk     = (const float*)d_in[2];
    const float* Wv     = (const float*)d_in[3];
    const float* E_w    = (const float*)d_in[4];
    const float* E_b    = (const float*)d_in[5];
    const float* F_w    = (const float*)d_in[6];
    const float* F_b    = (const float*)d_in[7];
    const float* Wo_w   = (const float*)d_in[8];
    const float* Wo_b   = (const float*)d_in[9];
    float*       out    = (float*)d_out;

    char* ws = (char*)d_ws;
    size_t off = 0;
    auto alloc = [&](size_t bytes) -> char* {
        char* p = ws + off;
        off = (off + bytes + 255) & ~(size_t)255;
        return p;
    };

    _Float16* Xh   = (_Float16*)alloc((size_t)Bn * S * Cn * 2);
    _Float16* Wqh  = (_Float16*)alloc((size_t)Hn * Dn * Cn * 2);
    _Float16* Wkh  = (_Float16*)alloc((size_t)Hn * Dn * Cn * 2);
    _Float16* Wvh  = (_Float16*)alloc((size_t)Hn * Dn * Cn * 2);
    _Float16* Ewh  = (_Float16*)alloc((size_t)Kn * S * 2);
    _Float16* Fwh  = (_Float16*)alloc((size_t)Kn * S * 2);
    _Float16* Woh  = (_Float16*)alloc((size_t)Cn * Hn * Dn * 2);
    _Float16* Q    = (_Float16*)alloc((size_t)Bn * Hn * S * Dn * 2);
    _Float16* Kt   = (_Float16*)alloc((size_t)Bn * Hn * S * Dn * 2);
    _Float16* V    = (_Float16*)alloc((size_t)Bn * Hn * S * Dn * 2);
    _Float16* KpT  = (_Float16*)alloc((size_t)Bn * Hn * Kn * Dn * 2);
    _Float16* Vp   = (_Float16*)alloc((size_t)Bn * Hn * Kn * Dn * 2);
    _Float16* SC   = (_Float16*)alloc((size_t)Bn * Hn * S * Kn * 2);   // scores -> P in place
    _Float16* GO   = (_Float16*)alloc((size_t)Bn * S * Hn * Dn * 2);   // gelu(concat) f16

    // ---- fp32 -> f16 conversions ----
    cvt_f32_to_f16<<<2048, 256, 0, stream>>>(tensor, Xh,  (long)Bn * S * Cn);
    cvt_f32_to_f16<<<1024, 256, 0, stream>>>(Wq,     Wqh, (long)Hn * Dn * Cn);
    cvt_f32_to_f16<<<1024, 256, 0, stream>>>(Wk,     Wkh, (long)Hn * Dn * Cn);
    cvt_f32_to_f16<<<1024, 256, 0, stream>>>(Wv,     Wvh, (long)Hn * Dn * Cn);
    cvt_f32_to_f16<<<1024, 256, 0, stream>>>(E_w,    Ewh, (long)Kn * S);
    cvt_f32_to_f16<<<1024, 256, 0, stream>>>(F_w,    Fwh, (long)Kn * S);
    cvt_f32_to_f16<<<1024, 256, 0, stream>>>(Wo_w,   Woh, (long)Cn * Hn * Dn);

    // ---- Q/K/V projections: X[b](SxC) . W[h]^T -> [b,h,S,D] ----
    {
        dim3 grid(1, S / TM, Bn * Hn);
        gemm_wmma_kernel<64, 0, 0, 0, _Float16><<<grid, 256, 0, stream>>>(
            Xh, Cn, (long)S * Cn, 0L,
            Wqh, Cn, 0L, (long)Dn * Cn,
            Q, Dn, (long)Hn * S * Dn, (long)S * Dn,
            nullptr, Cn, Hn);
        gemm_wmma_kernel<64, 0, 0, 0, _Float16><<<grid, 256, 0, stream>>>(
            Xh, Cn, (long)S * Cn, 0L,
            Wkh, Cn, 0L, (long)Dn * Cn,
            Kt, Dn, (long)Hn * S * Dn, (long)S * Dn,
            nullptr, Cn, Hn);
        gemm_wmma_kernel<64, 0, 0, 0, _Float16><<<grid, 256, 0, stream>>>(
            Xh, Cn, (long)S * Cn, 0L,
            Wvh, Cn, 0L, (long)Dn * Cn,
            V, Dn, (long)Hn * S * Dn, (long)S * Dn,
            nullptr, Cn, Hn);
    }

    // ---- Kp^T = E_w(KxS) . Kt(SxD) + E_b[k]; Vp = F_w . V + F_b[k] ----
    {
        dim3 grid(1, Kn / TM, Bn * Hn);
        gemm_wmma_kernel<64, 1, 1, 0, _Float16><<<grid, 256, 0, stream>>>(
            Ewh, S, 0L, 0L,
            Kt, Dn, (long)S * Dn, 0L,
            KpT, Dn, (long)Kn * Dn, 0L,
            E_b, S, 1);
        gemm_wmma_kernel<64, 1, 1, 0, _Float16><<<grid, 256, 0, stream>>>(
            Fwh, S, 0L, 0L,
            V, Dn, (long)S * Dn, 0L,
            Vp, Dn, (long)Kn * Dn, 0L,
            F_b, S, 1);
    }

    // ---- scores = Q(SxD) . KpT(KxD)^T -> [b,h,S,K] ----
    {
        dim3 grid(Kn / 128, S / TM, Bn * Hn);
        gemm_wmma_kernel<128, 0, 0, 0, _Float16><<<grid, 256, 0, stream>>>(
            Q, Dn, (long)S * Dn, 0L,
            KpT, Dn, (long)Kn * Dn, 0L,
            SC, Kn, (long)S * Kn, 0L,
            nullptr, Dn, 1);
    }

    // ---- softmax over K=256 with 1/sqrt(D) scale, in place ----
    {
        long rows = (long)Bn * Hn * S;
        softmax256_kernel<<<(unsigned)(rows / 8), 256, 0, stream>>>(SC, 0.125f);
    }

    // ---- out = P(SxK) . Vp(KxD), fused exact-GELU, write [b][s][h*D+d] ----
    {
        dim3 grid(1, S / TM, Bn * Hn);
        gemm_wmma_kernel<64, 1, 0, 1, _Float16><<<grid, 256, 0, stream>>>(
            SC, Kn, (long)Hn * S * Kn, (long)S * Kn,
            Vp, Dn, (long)Hn * Kn * Dn, (long)Kn * Dn,
            GO, Hn * Dn, (long)S * Hn * Dn, (long)Dn,
            nullptr, Kn, Hn);
    }

    // ---- final: GO(B*S x 1024) . Wo_w^T + Wo_b -> fp32 d_out ----
    {
        dim3 grid(Cn / 128, (Bn * S) / TM, 1);
        gemm_wmma_kernel<128, 0, 2, 0, float><<<grid, 256, 0, stream>>>(
            GO, Hn * Dn, 0L, 0L,
            Woh, Hn * Dn, 0L, 0L,
            out, Cn, 0L, 0L,
            Wo_b, Hn * Dn, 1);
    }
}
